// TraceNetLayer_37761352467017
// MI455X (gfx1250) — matrix operand order repaired and checked by reading, not compiled, not run on hardware
//
#include <hip/hip_runtime.h>

// ---------------- dims (fixed by reference) ----------------
#define BB 16
#define SS 2048
#define HH 1024
#define FF 1024
#define MM (BB * SS)   // 32768 rows
#define TM 64
#define TN 64
#define TK 32

typedef __attribute__((ext_vector_type(16))) __bf16 v16bf;
typedef __attribute__((ext_vector_type(8)))  __bf16 v8bf;
typedef __attribute__((ext_vector_type(8)))  float  v8f;

__device__ __forceinline__ __bf16 f2bf(float f) {
    union { float f; unsigned u; } a; a.f = f;
    unsigned u = a.u + 0x7FFFu + ((a.u >> 16) & 1u);   // round-to-nearest-even
    union { unsigned short s; __bf16 b; } r; r.s = (unsigned short)(u >> 16);
    return r.b;
}
__device__ __forceinline__ float bf2f(__bf16 b) {
    union { __bf16 b; unsigned short s; } u; u.b = b;
    union { unsigned u; float f; } r; r.u = ((unsigned)u.s) << 16;
    return r.f;
}

// CDNA5 async global->LDS copy (16B per lane), tracked with ASYNCcnt
__device__ __forceinline__ void async_copy16(unsigned lds_off, const void* gsrc) {
    asm volatile("global_load_async_to_lds_b128 %0, %1, off"
                 :: "v"(lds_off), "v"((unsigned long long)(uintptr_t)gsrc)
                 : "memory");
}
__device__ __forceinline__ void wait_async0() {
    asm volatile("s_wait_asynccnt 0x0" ::: "memory");
}

// A-fragment k-swizzle: within a 32-wide k row, swap groups [8..15] <-> [16..23]
// so a lane's 16 fragment elements {8g..8g+7, 16+8g..16+8g+7} become contiguous.
// (async chunks are 8 elements, matching the permutation groups exactly)
__device__ __forceinline__ int kswz(int k) {
    return (k & 7) | ((k & 8) << 1) | ((k & 16) >> 1);
}

#define WMMA_BF16(af, bf, acc) \
    __builtin_amdgcn_wmma_f32_16x16x32_bf16(false, (af), false, (bf), (short)0, (acc), false, false)

// =====================================================================
// flat convert X f32 -> bf16 (8 elems / thread)
// =====================================================================
__global__ void cvtX_k(const float* __restrict__ X, __bf16* __restrict__ Xb) {
    size_t i = ((size_t)blockIdx.x * 256 + threadIdx.x) * 8;
    float4 a = *(const float4*)&X[i];
    float4 b = *(const float4*)&X[i + 4];
    v8bf o;
    o[0] = f2bf(a.x); o[1] = f2bf(a.y); o[2] = f2bf(a.z); o[3] = f2bf(a.w);
    o[4] = f2bf(b.x); o[5] = f2bf(b.y); o[6] = f2bf(b.z); o[7] = f2bf(b.w);
    *(v8bf*)&Xb[i] = o;
}

// =====================================================================
// transpose + convert W[k][n] f32 -> Wt[n][k] bf16 (K = N = 1024)
// grid = (1024/32, 1024/32), block = 1024
// =====================================================================
__global__ __launch_bounds__(1024) void transcvt_k(const float* __restrict__ W,
                                                   __bf16* __restrict__ Wt) {
    __shared__ float tile[32][33];
    int tx = threadIdx.x & 31, ty = threadIdx.x >> 5;
    int kb = blockIdx.x * 32, nb = blockIdx.y * 32;
    tile[ty][tx] = W[(size_t)(kb + ty) * FF + nb + tx];
    __syncthreads();
    Wt[(size_t)(nb + ty) * FF + kb + tx] = f2bf(tile[tx][ty]);
}

// =====================================================================
// GEMM1: C = relu(Xb@WCt^T + bC) -> Cb (bf16);  A = relu(Xb@WAt^T + bA) -> Ab + A_out (f32)
// all-bf16 async double-buffered staging.  grid = (FF/TN, MM/TM), block = 128
// =====================================================================
__global__ __launch_bounds__(128) void gemm_proj_k(
    const __bf16* __restrict__ Xb, const __bf16* __restrict__ WCt,
    const __bf16* __restrict__ WAt,
    const float* __restrict__ bC, const float* __restrict__ bA,
    __bf16* __restrict__ Cb, __bf16* __restrict__ Ab, float* __restrict__ Aout)
{
    __shared__ __bf16 Xs[2][TM * TK];
    __shared__ __bf16 WCs[2][TN * TK];
    __shared__ __bf16 WAs[2][TN * TK];

    const int tid  = threadIdx.x;
    const int wid  = tid >> 5;
    const int lane = tid & 31;
    const int g    = lane >> 4;
    const int r16  = lane & 15;
    const int n0   = blockIdx.x * TN;
    const int m0   = blockIdx.y * TM;

    v8f accC[4] = {}, accA[4] = {};

    // per-thread 16B chunks: chunk c -> row=c>>2, k=(c&3)*8 ; thread does c=tid, tid+128
    const int row0 = tid >> 2, row1 = row0 + 32;
    const int kk   = (tid & 3) * 8;
    const int sk   = kk;                // unswizzled dest offset (contiguous k, same as B)

    #define ISSUE_PROJ(bufi, k0g) do {                                                            \
        async_copy16((unsigned)(uintptr_t)&Xs[bufi][row0 * TK + sk],                              \
                     &Xb[(size_t)(m0 + row0) * HH + (k0g) + kk]);                                 \
        async_copy16((unsigned)(uintptr_t)&Xs[bufi][row1 * TK + sk],                              \
                     &Xb[(size_t)(m0 + row1) * HH + (k0g) + kk]);                                 \
        async_copy16((unsigned)(uintptr_t)&WCs[bufi][row0 * TK + kk],                             \
                     &WCt[(size_t)(n0 + row0) * HH + (k0g) + kk]);                                \
        async_copy16((unsigned)(uintptr_t)&WCs[bufi][row1 * TK + kk],                             \
                     &WCt[(size_t)(n0 + row1) * HH + (k0g) + kk]);                                \
        async_copy16((unsigned)(uintptr_t)&WAs[bufi][row0 * TK + kk],                             \
                     &WAt[(size_t)(n0 + row0) * HH + (k0g) + kk]);                                \
        async_copy16((unsigned)(uintptr_t)&WAs[bufi][row1 * TK + kk],                             \
                     &WAt[(size_t)(n0 + row1) * HH + (k0g) + kk]);                                \
    } while (0)

    ISSUE_PROJ(0, 0);
    const int NT = HH / TK;
    for (int t = 0; t < NT; ++t) {
        const int buf = t & 1;
        wait_async0();          // tile t resident in LDS
        __syncthreads();        // + all waves done reading buf^1
        if (t + 1 < NT) ISSUE_PROJ(buf ^ 1, (t + 1) * TK);

        v16bf af = *(const v16bf*)&Xs[buf][(wid * 16 + r16) * TK + g * 16];
        v16bf bc[4], ba[4];
        #pragma unroll
        for (int nt = 0; nt < 4; ++nt) {
            bc[nt] = *(const v16bf*)&WCs[buf][(nt * 16 + r16) * TK + g * 16];
            ba[nt] = *(const v16bf*)&WAs[buf][(nt * 16 + r16) * TK + g * 16];
        }
        #pragma unroll
        for (int nt = 0; nt < 4; ++nt) {
            accC[nt] = WMMA_BF16(af, bc[nt], accC[nt]);
            accA[nt] = WMMA_BF16(af, ba[nt], accA[nt]);
        }
    }
    #undef ISSUE_PROJ

    // epilogue: bias + relu; store bf16 (C,A) and f32 (A output)
    #pragma unroll
    for (int nt = 0; nt < 4; ++nt) {
        int n = n0 + nt * 16 + r16;
        float biasC = bC[n], biasA = bA[n];
        #pragma unroll
        for (int r = 0; r < 8; ++r) {
            int row = m0 + wid * 16 + g * 8 + r;
            size_t o = (size_t)row * FF + n;
            float c = accC[nt][r] + biasC; c = c > 0.f ? c : 0.f;
            float a = accA[nt][r] + biasA; a = a > 0.f ? a : 0.f;
            Cb[o] = f2bf(c);
            Ab[o] = f2bf(a);
            Aout[o] = a;
        }
    }
}

// =====================================================================
// GEMM2 fused: partial[nblk][row] = sum_{n in tile} tanh((Mb@w2)[row,n] + add[b,n]) * v[n]
// async double-buffered staging.  grid = (FF/TN, MM/TM), block = 128
// =====================================================================
__global__ __launch_bounds__(128) void gemm_attn_k(
    const __bf16* __restrict__ Mb, const __bf16* __restrict__ W2t,
    const float* __restrict__ addv /*[B][F] = qw1+attn_bias*/,
    const float* __restrict__ vvec, float* __restrict__ partial /*[FF/TN][MM]*/)
{
    __shared__ __bf16 Ms[2][TM * TK];
    __shared__ __bf16 Ws[2][TN * TK];

    const int tid  = threadIdx.x;
    const int wid  = tid >> 5;
    const int lane = tid & 31;
    const int g    = lane >> 4;
    const int r16  = lane & 15;
    const int n0   = blockIdx.x * TN;
    const int m0   = blockIdx.y * TM;
    const int b    = m0 / SS;               // TM divides SS

    v8f acc[4] = {};

    const int row0 = tid >> 2, row1 = row0 + 32;
    const int kk   = (tid & 3) * 8;
    const int sk   = kk;

    #define ISSUE_ATTN(bufi, k0g) do {                                                            \
        async_copy16((unsigned)(uintptr_t)&Ms[bufi][row0 * TK + sk],                              \
                     &Mb[(size_t)(m0 + row0) * FF + (k0g) + kk]);                                 \
        async_copy16((unsigned)(uintptr_t)&Ms[bufi][row1 * TK + sk],                              \
                     &Mb[(size_t)(m0 + row1) * FF + (k0g) + kk]);                                 \
        async_copy16((unsigned)(uintptr_t)&Ws[bufi][row0 * TK + kk],                              \
                     &W2t[(size_t)(n0 + row0) * FF + (k0g) + kk]);                                \
        async_copy16((unsigned)(uintptr_t)&Ws[bufi][row1 * TK + kk],                              \
                     &W2t[(size_t)(n0 + row1) * FF + (k0g) + kk]);                                \
    } while (0)

    ISSUE_ATTN(0, 0);
    const int NT = FF / TK;
    for (int t = 0; t < NT; ++t) {
        const int buf = t & 1;
        wait_async0();
        __syncthreads();
        if (t + 1 < NT) ISSUE_ATTN(buf ^ 1, (t + 1) * TK);

        v16bf af = *(const v16bf*)&Ms[buf][(wid * 16 + r16) * TK + g * 16];
        v16bf bfr[4];
        #pragma unroll
        for (int nt = 0; nt < 4; ++nt)
            bfr[nt] = *(const v16bf*)&Ws[buf][(nt * 16 + r16) * TK + g * 16];
        #pragma unroll
        for (int nt = 0; nt < 4; ++nt)
            acc[nt] = WMMA_BF16(af, bfr[nt], acc[nt]);
    }
    #undef ISSUE_ATTN

    // epilogue: e = tanh(acc + add[b,n]) * v[n]; reduce over the 64 columns
    float rowsum[8];
    #pragma unroll
    for (int r = 0; r < 8; ++r) rowsum[r] = 0.f;
    #pragma unroll
    for (int nt = 0; nt < 4; ++nt) {
        int n = n0 + nt * 16 + r16;
        float ad = addv[b * FF + n];
        float vv = vvec[n];
        #pragma unroll
        for (int r = 0; r < 8; ++r)
            rowsum[r] += tanhf(acc[nt][r] + ad) * vv;
    }
    #pragma unroll
    for (int r = 0; r < 8; ++r) {
        float s = rowsum[r];
        s += __shfl_xor(s, 1, 32);
        s += __shfl_xor(s, 2, 32);
        s += __shfl_xor(s, 4, 32);
        s += __shfl_xor(s, 8, 32);
        if (r16 == 0) {
            int row = m0 + wid * 16 + g * 8 + r;
            partial[(size_t)blockIdx.x * MM + row] = s;
        }
    }
}

// generic deterministic partial-sum reduce: out[i] = sum_{j<count} partial[j*n+i]
__global__ void reduceN_k(const float* __restrict__ partial, float* __restrict__ out,
                          int n, int count) {
    int i = blockIdx.x * 256 + threadIdx.x;
    if (i < n) {
        float s = 0.f;
        for (int j = 0; j < count; ++j) s += partial[(size_t)j * n + i];
        out[i] = s;
    }
}
// reduce + per-column bias (for q@w1 + attn_bias)
__global__ void reduce_bias_k(const float* __restrict__ partial, const float* __restrict__ bias,
                              float* __restrict__ out, int n, int count) {
    int i = blockIdx.x * 256 + threadIdx.x;
    if (i < n) {
        float s = bias[i & (FF - 1)];
        for (int j = 0; j < count; ++j) s += partial[(size_t)j * n + i];
        out[i] = s;
    }
}

// partial[c][b*F+f] = sum_{s in chunk c} bf2f(Cb[b,s,f]) * w[b*S+s]
// grid = (16, B), block = 128 (each thread: 8 consecutive f via v8bf)
__global__ __launch_bounds__(128) void wsum_k(const __bf16* __restrict__ Cb,
                                              const float* __restrict__ w,
                                              float* __restrict__ partial) {
    int b = blockIdx.y, c = blockIdx.x, tid = threadIdx.x;
    float acc[8] = {};
    int s0 = c * (SS / 16);
    for (int s = s0; s < s0 + SS / 16; ++s) {
        float ww = w[b * SS + s];
        v8bf cv = *(const v8bf*)&Cb[((size_t)(b * SS + s)) * FF + tid * 8];
        #pragma unroll
        for (int i = 0; i < 8; ++i) acc[i] += bf2f(cv[i]) * ww;
    }
    #pragma unroll
    for (int i = 0; i < 8; ++i)
        partial[(size_t)c * (BB * FF) + b * FF + tid * 8 + i] = acc[i];
}

// partial[cz][b*F+n] = sum_{f in chunk} q[b,f]*w1[f,n];  grid (F/256, B, 4)
__global__ void smallgemm_k(const float* __restrict__ q, const float* __restrict__ w1,
                            float* __restrict__ partial) {
    int b = blockIdx.y, n = blockIdx.x * 256 + threadIdx.x, cz = blockIdx.z;
    float acc = 0.f;
    int f0 = cz * 256;
    for (int f = f0; f < f0 + 256; ++f)
        acc += q[b * FF + f] * w1[(size_t)f * FF + n];
    partial[(size_t)cz * (BB * FF) + b * FF + n] = acc;
}

// softmax over S per batch, with mask_add folded in
__global__ __launch_bounds__(256) void softmax_k(const float* __restrict__ logits,
                                                 const float* __restrict__ amask,
                                                 float* __restrict__ alpha) {
    int b = blockIdx.x, tid = threadIdx.x;
    __shared__ float red[256];
    float m = -1e30f;
    for (int s = tid; s < SS; s += 256) {
        float v = logits[b * SS + s] + (1.f - amask[b * SS + s]) * -10000.f;
        m = fmaxf(m, v);
    }
    red[tid] = m; __syncthreads();
    for (int o = 128; o > 0; o >>= 1) { if (tid < o) red[tid] = fmaxf(red[tid], red[tid + o]); __syncthreads(); }
    m = red[0]; __syncthreads();
    float sum = 0.f;
    for (int s = tid; s < SS; s += 256) {
        float v = logits[b * SS + s] + (1.f - amask[b * SS + s]) * -10000.f;
        sum += expf(v - m);
    }
    red[tid] = sum; __syncthreads();
    for (int o = 128; o > 0; o >>= 1) { if (tid < o) red[tid] += red[tid + o]; __syncthreads(); }
    float inv = 1.f / red[0];
    for (int s = tid; s < SS; s += 256) {
        float v = logits[b * SS + s] + (1.f - amask[b * SS + s]) * -10000.f;
        alpha[b * SS + s] = expf(v - m) * inv;
    }
}

// sparsemax over S per batch: bitonic sort (desc) + scan in LDS
__global__ __launch_bounds__(256) void sparsemax_k(
    const float* __restrict__ Lraw, const float* __restrict__ amask,
    const float* __restrict__ w_1, const float* __restrict__ w_2,
    const float* __restrict__ w_3, const int* __restrict__ layer_i,
    float* __restrict__ Lout)
{
    int b = blockIdx.x, tid = threadIdx.x;
    __shared__ float za[SS];
    __shared__ float zb[SS];
    __shared__ float zc[SS];
    __shared__ int kcnt;
    __shared__ float taus;

    float a1 = w_1[0], a2 = w_2[0], a3 = w_3[0];
    int li = layer_i[0];
    float x = (li == 0) ? (a3 * a3 - a2 * a2 - a1 * a1)
            : (li == 1) ? (a3 * a3 - a2 * a2) : (a3 * a3);
    float sg = 1.f / (1.f + expf(-x));

    for (int s = tid; s < SS; s += 256) {
        float v = Lraw[b * SS + s] * sg + (1.f - amask[b * SS + s]) * -10000.f;
        za[s] = v; zc[s] = v;
    }
    __syncthreads();

    // bitonic sort descending
    for (int k = 2; k <= SS; k <<= 1) {
        for (int j = k >> 1; j > 0; j >>= 1) {
            for (int i = tid; i < SS; i += 256) {
                int ixj = i ^ j;
                if (ixj > i) {
                    bool dir = ((i & k) == 0);
                    float v0 = za[i], v1 = za[ixj];
                    if ((v0 < v1) == dir) { za[i] = v1; za[ixj] = v0; }
                }
            }
            __syncthreads();
        }
    }

    // inclusive scan (Hillis-Steele, ping-pong)
    float* src = za; float* dst = zb;
    for (int off = 1; off < SS; off <<= 1) {
        for (int i = tid; i < SS; i += 256)
            dst[i] = src[i] + (i >= off ? src[i - off] : 0.f);
        __syncthreads();
        float* t = src; src = dst; dst = t;
    }

    if (tid == 0) kcnt = 0;
    __syncthreads();
    int localc = 0;
    for (int i = tid; i < SS; i += 256) {
        float cum = src[i];
        float zi = (i == 0) ? cum : cum - src[i - 1];   // recover sorted value
        if (1.f + (float)(i + 1) * zi > cum) localc++;
    }
    atomicAdd(&kcnt, localc);
    __syncthreads();
    if (tid == 0) taus = (src[kcnt - 1] - 1.f) / (float)kcnt;
    __syncthreads();
    float tau = taus;
    for (int s = tid; s < SS; s += 256) {
        float v = zc[s] - tau;
        Lout[b * SS + s] = v > 0.f ? v : 0.f;
    }
}

// =====================================================================
extern "C" void kernel_launch(void* const* d_in, const int* in_sizes, int n_in,
                              void* d_out, int out_size, void* d_ws, size_t ws_size,
                              hipStream_t stream)
{
    const float* amask = (const float*)d_in[0];
    const float* X     = (const float*)d_in[1];
    const float* iw    = (const float*)d_in[2];
    const float* WC    = (const float*)d_in[3];
    const float* bC    = (const float*)d_in[4];
    const float* WA    = (const float*)d_in[5];
    const float* bA    = (const float*)d_in[6];
    const float* w1    = (const float*)d_in[7];
    const float* w2    = (const float*)d_in[8];
    const float* abias = (const float*)d_in[9];
    const float* vvec  = (const float*)d_in[10];
    const float* w_1   = (const float*)d_in[11];
    const float* w_2   = (const float*)d_in[12];
    const float* w_3   = (const float*)d_in[13];
    const int*  layeri = (const int*)d_in[14];

    float* out   = (float*)d_out;
    float* h_out = out;                       // [B*F]
    float* L_out = out + BB * FF;             // [B*S]
    float* A_out = out + BB * FF + BB * SS;   // [B*S*F]

    char* ws = (char*)d_ws;
    __bf16* Xb  = (__bf16*)ws;  ws += (size_t)MM * HH * 2;
    __bf16* Cb  = (__bf16*)ws;  ws += (size_t)MM * FF * 2;
    __bf16* Ab  = (__bf16*)ws;  ws += (size_t)MM * FF * 2;
    __bf16* WCt = (__bf16*)ws;  ws += (size_t)HH * FF * 2;
    __bf16* WAt = (__bf16*)ws;  ws += (size_t)HH * FF * 2;
    __bf16* W2t = (__bf16*)ws;  ws += (size_t)FF * FF * 2;
    float* qv     = (float*)ws; ws += (size_t)BB * FF * 4;
    float* qw1v   = (float*)ws; ws += (size_t)BB * FF * 4;
    float* logits = (float*)ws; ws += (size_t)MM * 4;
    float* alpha  = (float*)ws; ws += (size_t)MM * 4;
    float* hw1v   = (float*)ws; ws += (size_t)BB * FF * 4;
    float* Lraw   = (float*)ws; ws += (size_t)MM * 4;
    float* partial= (float*)ws; ws += (size_t)(FF / TN) * MM * 4;   // GEMM2 partials
    float* wpart  = (float*)ws; ws += (size_t)16 * BB * FF * 4;     // wsum partials
    float* spart  = (float*)ws; ws += (size_t)4 * BB * FF * 4;      // smallgemm partials

    dim3 gblk(128);
    dim3 ggrid(FF / TN, MM / TM);   // (16, 512)
    dim3 tgrid(HH / 32, FF / 32);   // weight transpose grid

    // one-time conversions: X flat, weights transposed bf16 (enables pure async staging)
    cvtX_k<<<(MM * HH) / (8 * 256), 256, 0, stream>>>(X, Xb);
    transcvt_k<<<tgrid, 1024, 0, stream>>>(WC, WCt);
    transcvt_k<<<tgrid, 1024, 0, stream>>>(WA, WAt);
    transcvt_k<<<tgrid, 1024, 0, stream>>>(w2, W2t);
    // projections: C (bf16) and A (bf16 + f32 output)
    gemm_proj_k<<<ggrid, gblk, 0, stream>>>(Xb, WCt, WAt, bC, bA, Cb, Ab, A_out);
    // q = sum_s C * item_weights
    wsum_k<<<dim3(16, BB), 128, 0, stream>>>(Cb, iw, wpart);
    reduceN_k<<<(BB * FF) / 256, 256, 0, stream>>>(wpart, qv, BB * FF, 16);
    // qw1 = q@w1 + attn_bias
    smallgemm_k<<<dim3(FF / 256, BB, 4), 256, 0, stream>>>(qv, w1, spart);
    reduce_bias_k<<<(BB * FF) / 256, 256, 0, stream>>>(spart, abias, qw1v, BB * FF, 4);
    // alpha logits = (tanh(C@w2 + qw1 + bias)) @ v
    gemm_attn_k<<<ggrid, gblk, 0, stream>>>(Cb, W2t, qw1v, vvec, partial);
    reduceN_k<<<MM / 256, 256, 0, stream>>>(partial, logits, MM, 16);
    softmax_k<<<BB, 256, 0, stream>>>(logits, amask, alpha);
    // h = sum_s C * alpha  (straight to output)
    wsum_k<<<dim3(16, BB), 128, 0, stream>>>(Cb, alpha, wpart);
    reduceN_k<<<(BB * FF) / 256, 256, 0, stream>>>(wpart, h_out, BB * FF, 16);
    // hw1 = h@w1 + attn_bias
    smallgemm_k<<<dim3(FF / 256, BB, 4), 256, 0, stream>>>(h_out, w1, spart);
    reduce_bias_k<<<(BB * FF) / 256, 256, 0, stream>>>(spart, abias, hw1v, BB * FF, 4);
    // L raw = (tanh(A@w2 + hw1 + bias)) @ v
    gemm_attn_k<<<ggrid, gblk, 0, stream>>>(Ab, W2t, hw1v, vvec, partial);
    reduceN_k<<<MM / 256, 256, 0, stream>>>(partial, Lraw, MM, 16);
    // L = sparsemax(L*sigmoid(x) + mask_add)
    sparsemax_k<<<BB, 256, 0, stream>>>(Lraw, amask, w_1, w_2, w_3, layeri, L_out);
}
